// LatentSelfAttention_82566451298777
// MI455X (gfx1250) — matrix operand order repaired and checked by reading, not compiled, not run on hardware
//
#include <hip/hip_runtime.h>

// ---------------------------------------------------------------------------
// LatentSelfAttention for MI455X (gfx1250, wave32, WMMA bf16)
//   x[4,4096,2048] f32; Wq/Wk/Wv[2048,512]; Wo[512,2048]; out f32 [4,4096,2048]
// Pipeline: cvt x->bf16, transpose weights->bf16, QKV WMMA GEMMs,
//           flash-attention (async-LDS staged, online softmax), output GEMM.
// ---------------------------------------------------------------------------

#define BATCH 4
#define TSEQ  4096
#define DMODEL 2048
#define LAT   512
#define MROWS (BATCH * TSEQ)   // 16384

typedef __attribute__((ext_vector_type(16))) __bf16 bf16x16;
typedef __attribute__((ext_vector_type(8)))  float  f32x8;

union Frag {
    bf16x16 f;
    uint4   u[2];
};

static __device__ __forceinline__ unsigned short f2bf(float x) {
    unsigned u = __builtin_bit_cast(unsigned, x);
    // round-to-nearest-even truncation to bf16
    unsigned r = (u + 0x7FFFu + ((u >> 16) & 1u)) >> 16;
    return (unsigned short)r;
}

static __device__ __forceinline__ f32x8 wmma_bf16(const Frag& a, const Frag& b, f32x8 c) {
    // (neg_a, A, neg_b, B, c_mod, C, reuse_a, reuse_b)
    return __builtin_amdgcn_wmma_f32_16x16x32_bf16(false, a.f, false, b.f,
                                                   (short)0, c, false, false);
}

// ---- gfx1250 async global->LDS copy (ASYNCcnt-tracked DMA, 16B per lane) ----
// Flat LDS pointers carry the byte offset in addr[31:0] (ISA 10.2 aperture
// decode), so truncation gives the LDS-relative address the instruction wants.
static __device__ __forceinline__ void async_ld_b128(void* lds_ptr, const void* gptr) {
    unsigned lds_off = (unsigned)(unsigned long long)lds_ptr;
    unsigned long long ga = (unsigned long long)gptr;
    asm volatile("global_load_async_to_lds_b128 %0, %1, off"
                 :: "v"(lds_off), "v"(ga)
                 : "memory");
}
static __device__ __forceinline__ void wait_async0() {
    asm volatile("s_wait_asynccnt 0" ::: "memory");
}

// ---------------------------------------------------------------------------
// 1) f32 -> bf16 bulk convert (8 elems / thread, b128 in, b128 out)
// ---------------------------------------------------------------------------
__global__ __launch_bounds__(256) void cvt_bf16_kernel(const float* __restrict__ in,
                                                       unsigned short* __restrict__ out) {
    size_t i = ((size_t)blockIdx.x * 256 + threadIdx.x) * 8;
    float4 a = *(const float4*)(in + i);
    float4 b = *(const float4*)(in + i + 4);
    uint4 o;
    o.x = (unsigned)f2bf(a.x) | ((unsigned)f2bf(a.y) << 16);
    o.y = (unsigned)f2bf(a.z) | ((unsigned)f2bf(a.w) << 16);
    o.z = (unsigned)f2bf(b.x) | ((unsigned)f2bf(b.y) << 16);
    o.w = (unsigned)f2bf(b.z) | ((unsigned)f2bf(b.w) << 16);
    *(uint4*)(out + i) = o;
}

// ---------------------------------------------------------------------------
// 2) transpose f32 [R][C] -> bf16 [C][R] (weights are tiny; simple is fine)
// ---------------------------------------------------------------------------
__global__ __launch_bounds__(256) void transpose_bf16_kernel(const float* __restrict__ in,
                                                             unsigned short* __restrict__ out,
                                                             int R, int C) {
    int idx = blockIdx.x * 256 + threadIdx.x;
    if (idx < R * C) {
        int r = idx / C, c = idx % C;
        out[(size_t)c * R + r] = f2bf(in[idx]);
    }
}

// ---------------------------------------------------------------------------
// 3/5) WMMA GEMM: C[M,N] = A[M,KD](bf16, k-contig) * Bt[N,KD](bf16, k-contig)^T
//      workgroup tile 64x64, 4 waves * 16 rows, K-step 32.
//      A-frag: lane(m)=lane&15, k chunks {hi*16 .. +15}  (two b128)
//      B-frag: lane(n)=lane&15, k chunks {hi*16 .. +15}  (two b128)
// ---------------------------------------------------------------------------
template <int KD, bool OUT_BF16>
__global__ __launch_bounds__(128) void gemm_kernel(const unsigned short* __restrict__ A,
                                                   const unsigned short* __restrict__ Bt,
                                                   void* __restrict__ Cv, int N) {
    const int lane = threadIdx.x & 31;
    const int wave = threadIdx.x >> 5;
    const int ln = lane & 15;
    const int hi = lane >> 4;

    const int m0 = blockIdx.x * 64 + wave * 16;
    const int n0 = blockIdx.y * 64;

    f32x8 acc[4];
#pragma unroll
    for (int t = 0; t < 4; ++t) acc[t] = (f32x8){};

    const unsigned short* arow = A + (size_t)(m0 + ln) * KD + hi * 16;

    for (int k = 0; k < KD; k += 32) {
        __builtin_prefetch(arow + k + 128, 0, 1);   // global_prefetch_b8
        Frag a;
        const uint4* ap = (const uint4*)(arow + k);
        a.u[0] = ap[0];
        a.u[1] = ap[1];   // +8 bf16 elements
#pragma unroll
        for (int nt = 0; nt < 4; ++nt) {
            Frag b;
            const uint4* bp = (const uint4*)(Bt + (size_t)(n0 + nt * 16 + ln) * KD + k + hi * 16);
            b.u[0] = bp[0];
            b.u[1] = bp[1];
            acc[nt] = wmma_bf16(a, b, acc[nt]);
        }
    }

#pragma unroll
    for (int nt = 0; nt < 4; ++nt) {
#pragma unroll
        for (int r = 0; r < 8; ++r) {
            int row = m0 + hi * 8 + r;
            int col = n0 + nt * 16 + ln;
            if (OUT_BF16)
                ((unsigned short*)Cv)[(size_t)row * N + col] = f2bf(acc[nt][r]);
            else
                ((float*)Cv)[(size_t)row * N + col] = acc[nt][r];
        }
    }
}

// ---------------------------------------------------------------------------
// 4) Flash attention, causal. Grid (T/64, B), 128 threads (4 waves).
//    Each wave: 16 query rows, O accumulator 16x512 f32 (32 WMMA C tiles).
//    Q and K blocks staged with GLOBAL_LOAD_ASYNC_TO_LDS_B128 (ASYNCcnt DMA);
//    V staged transposed via VGPR scatter (layout change needed for PV B-op).
// ---------------------------------------------------------------------------
__global__ __launch_bounds__(128) void attn_kernel(const unsigned short* __restrict__ Q,
                                                   const unsigned short* __restrict__ K,
                                                   const unsigned short* __restrict__ V,
                                                   unsigned short* __restrict__ O) {
    __shared__ unsigned short q_lds[64 * LAT];    // 64 KB: query block, row-major
    __shared__ unsigned short k_lds[32 * LAT];    // 32 KB: key block, row-major
    __shared__ unsigned short vt_lds[LAT * 32];   // 32 KB: value block transposed [l][s]
    __shared__ unsigned short p_lds[4 * 16 * 32]; //  4 KB: per-wave P tile

    const int b = blockIdx.y;
    const int q0 = blockIdx.x * 64;
    const int tid = threadIdx.x;
    const int wave = tid >> 5;
    const int lane = tid & 31;
    const int ln = lane & 15;
    const int hi = lane >> 4;

    const unsigned short* Qb = Q + (size_t)b * TSEQ * LAT;
    const unsigned short* Kb = K + (size_t)b * TSEQ * LAT;
    const unsigned short* Vb = V + (size_t)b * TSEQ * LAT;
    unsigned short* Ob = O + (size_t)b * TSEQ * LAT;

    // stage Q block (64x512 bf16 = 4096 x 16B) via async DMA to LDS
    {
        const unsigned short* src = Qb + (size_t)q0 * LAT;
        for (int i = tid; i < 4096; i += 128)
            async_ld_b128(&q_lds[i * 8], src + (size_t)i * 8);
    }
    wait_async0();
    __syncthreads();

    f32x8 o_acc[32];
#pragma unroll
    for (int t = 0; t < 32; ++t) o_acc[t] = (f32x8){};
    float m_i[8], l_i[8];
#pragma unroll
    for (int r = 0; r < 8; ++r) { m_i[r] = -1e30f; l_i[r] = 0.0f; }

    const float scale = 0.04419417382415922f; // 1/sqrt(512)
    const int row_base = q0 + wave * 16 + hi * 8;

    for (int s0 = 0; s0 <= q0 + 63; s0 += 32) {
        __syncthreads();  // previous iteration's LDS reads done before restaging
        // stage K block rows [s0, s0+32): async DMA, 16 x 16B per thread
        {
            const unsigned short* src = Kb + (size_t)s0 * LAT;
            for (int i = tid; i < 2048; i += 128)
                async_ld_b128(&k_lds[i * 8], src + (size_t)i * 8);
        }
        // stage V block transposed: vt_lds[l][s] = V[s0+s][l]
        {
            const uint4* src = (const uint4*)(Vb + (size_t)s0 * LAT);
            for (int i = tid; i < 2048; i += 128) {
                uint4 d = src[i];
                int s = i >> 6;              // row within block (64 uint4 per row)
                int c0 = (i & 63) << 3;      // starting latent column
                const unsigned short* e = (const unsigned short*)&d;
#pragma unroll
                for (int j = 0; j < 8; ++j) vt_lds[(c0 + j) * 32 + s] = e[j];
            }
        }
        wait_async0();
        __syncthreads();

        // ---- S = Q K^T over the 512-deep contraction ----
        f32x8 sc0 = (f32x8){}, sc1 = (f32x8){};
#pragma unroll
        for (int lk = 0; lk < LAT; lk += 32) {
            Frag a, b0, b1;
            const uint4* qa = (const uint4*)(q_lds + (wave * 16 + ln) * LAT + lk + hi * 16);
            a.u[0] = qa[0];
            a.u[1] = qa[1];
            const uint4* kb0 = (const uint4*)(k_lds + ln * LAT + lk + hi * 16);
            b0.u[0] = kb0[0];
            b0.u[1] = kb0[1];
            const uint4* kb1 = (const uint4*)(k_lds + (16 + ln) * LAT + lk + hi * 16);
            b1.u[0] = kb1[0];
            b1.u[1] = kb1[1];
            sc0 = wmma_bf16(a, b0, sc0);
            sc1 = wmma_bf16(a, b1, sc1);
        }

        // ---- online softmax update (row stats live in 16-lane halves) ----
#pragma unroll
        for (int r = 0; r < 8; ++r) {
            const int row = row_base + r;
            float s0v = sc0[r] * scale;
            float s1v = sc1[r] * scale;
            const int col0 = s0 + ln;
            const int col1 = s0 + 16 + ln;
            if (col0 > row) s0v = -1e30f;
            if (col1 > row) s1v = -1e30f;
            float rmax = fmaxf(s0v, s1v);
            rmax = fmaxf(rmax, __shfl_xor(rmax, 1));
            rmax = fmaxf(rmax, __shfl_xor(rmax, 2));
            rmax = fmaxf(rmax, __shfl_xor(rmax, 4));
            rmax = fmaxf(rmax, __shfl_xor(rmax, 8));
            const float mnew = fmaxf(m_i[r], rmax);
            const float corr = __expf(m_i[r] - mnew);
            const float p0 = (s0v < -1e29f) ? 0.0f : __expf(s0v - mnew);
            const float p1 = (s1v < -1e29f) ? 0.0f : __expf(s1v - mnew);
            float rsum = p0 + p1;
            rsum += __shfl_xor(rsum, 1);
            rsum += __shfl_xor(rsum, 2);
            rsum += __shfl_xor(rsum, 4);
            rsum += __shfl_xor(rsum, 8);
            l_i[r] = l_i[r] * corr + rsum;
            m_i[r] = mnew;
#pragma unroll
            for (int lt = 0; lt < 32; ++lt) o_acc[lt][r] *= corr;
            // C-layout f32 -> row-major bf16 staging of P
            p_lds[wave * 512 + (hi * 8 + r) * 32 + ln] = f2bf(p0);
            p_lds[wave * 512 + (hi * 8 + r) * 32 + 16 + ln] = f2bf(p1);
        }

        // ---- O += P V  (A-layout P from LDS, B = V^T from LDS) ----
        Frag pa;
        const uint4* pp = (const uint4*)(p_lds + wave * 512 + ln * 32 + hi * 16);
        pa.u[0] = pp[0];
        pa.u[1] = pp[1];
#pragma unroll
        for (int lt = 0; lt < 32; ++lt) {
            Frag vb;
            const uint4* vv = (const uint4*)(vt_lds + (lt * 16 + ln) * 32 + hi * 16);
            vb.u[0] = vv[0];
            vb.u[1] = vv[1];
            o_acc[lt] = wmma_bf16(pa, vb, o_acc[lt]);
        }
    }

    // epilogue: normalize rows, emit bf16 latent output for the final GEMM
#pragma unroll
    for (int r = 0; r < 8; ++r) {
        const float inv = 1.0f / l_i[r];
        const int row = row_base + r;
#pragma unroll
        for (int lt = 0; lt < 32; ++lt) {
            Ob[(size_t)row * LAT + lt * 16 + ln] = f2bf(o_acc[lt][r] * inv);
        }
    }
}

// ---------------------------------------------------------------------------
// host-side orchestration
// ---------------------------------------------------------------------------
extern "C" void kernel_launch(void* const* d_in, const int* in_sizes, int n_in,
                              void* d_out, int out_size, void* d_ws, size_t ws_size,
                              hipStream_t stream) {
    (void)in_sizes; (void)n_in; (void)out_size; (void)ws_size;
    const float* x  = (const float*)d_in[0];
    const float* Wq = (const float*)d_in[1];
    const float* Wk = (const float*)d_in[2];
    const float* Wv = (const float*)d_in[3];
    const float* Wo = (const float*)d_in[4];
    float* out = (float*)d_out;

    char* ws = (char*)d_ws;
    size_t off = 0;
    unsigned short* xb  = (unsigned short*)(ws + off); off += (size_t)MROWS * DMODEL * 2;  // 64 MB
    unsigned short* Wqt = (unsigned short*)(ws + off); off += (size_t)DMODEL * LAT * 2;
    unsigned short* Wkt = (unsigned short*)(ws + off); off += (size_t)DMODEL * LAT * 2;
    unsigned short* Wvt = (unsigned short*)(ws + off); off += (size_t)DMODEL * LAT * 2;
    unsigned short* Wot = (unsigned short*)(ws + off); off += (size_t)DMODEL * LAT * 2;
    unsigned short* Qw  = (unsigned short*)(ws + off); off += (size_t)MROWS * LAT * 2;
    unsigned short* Kw  = (unsigned short*)(ws + off); off += (size_t)MROWS * LAT * 2;
    unsigned short* Vw  = (unsigned short*)(ws + off); off += (size_t)MROWS * LAT * 2;
    unsigned short* Ol  = (unsigned short*)(ws + off); off += (size_t)MROWS * LAT * 2;

    // 1) x -> bf16 (33.5M elems, 8/thread)
    cvt_bf16_kernel<<<(MROWS * (size_t)DMODEL) / (256 * 8), 256, 0, stream>>>(x, xb);

    // 2) weight transposes to [n][k] bf16
    {
        int nElem = DMODEL * LAT;
        int blocks = (nElem + 255) / 256;
        transpose_bf16_kernel<<<blocks, 256, 0, stream>>>(Wq, Wqt, DMODEL, LAT);
        transpose_bf16_kernel<<<blocks, 256, 0, stream>>>(Wk, Wkt, DMODEL, LAT);
        transpose_bf16_kernel<<<blocks, 256, 0, stream>>>(Wv, Wvt, DMODEL, LAT);
        transpose_bf16_kernel<<<blocks, 256, 0, stream>>>(Wo, Wot, LAT, DMODEL);
    }

    // 3) Q/K/V projections: [16384,2048] x [2048,512] -> bf16
    {
        dim3 grid(MROWS / 64, LAT / 64);
        gemm_kernel<DMODEL, true><<<grid, 128, 0, stream>>>(xb, Wqt, Qw, LAT);
        gemm_kernel<DMODEL, true><<<grid, 128, 0, stream>>>(xb, Wkt, Kw, LAT);
        gemm_kernel<DMODEL, true><<<grid, 128, 0, stream>>>(xb, Wvt, Vw, LAT);
    }

    // 4) causal flash attention -> latent output bf16
    attn_kernel<<<dim3(TSEQ / 64, BATCH), 128, 0, stream>>>(Qw, Kw, Vw, Ol);

    // 5) output projection: [16384,512] x [512,2048] -> f32 d_out
    gemm_kernel<LAT, false><<<dim3(MROWS / 64, DMODEL / 64), 128, 0, stream>>>(Ol, Wot, out, DMODEL);
}
